// ExtractiveLongformerForCache_35321811042493
// MI455X (gfx1250) — matrix operand order repaired and checked by reading, compiled, not run on hardware
//
#include <hip/hip_runtime.h>
#include <hip/hip_bf16.h>
#include <math.h>

// ---------------------------------------------------------------------------
// Longformer-style encoder forward on gfx1250 (MI455X), WMMA bf16 GEMM core.
// B=1, S=2048, HID=768, NH=12, DH=64, NL=4, WIN=512 (W1=256), G=64, FFN=3072
// ---------------------------------------------------------------------------

#define S_    2048
#define HID_  768
#define NH_   12
#define DH_   64
#define NL_   4
#define W1_   256
#define G_    64
#define FFN_  3072
#define BAND_ (2 * W1_ + 1)   // 513
#define NKEY_ (G_ + BAND_)    // 577
#define NEGV_ (-1.0e9f)
#define EPS_  1e-5f
#define NR_   31              // rows in the (31,63) post slice

typedef __attribute__((ext_vector_type(16))) __bf16 v16bf;
typedef __attribute__((ext_vector_type(8)))  __bf16 v8bf;
typedef __attribute__((ext_vector_type(8)))  float  v8f;

__device__ __forceinline__ __bf16 f2bf(float f) {
  unsigned u = __float_as_uint(f);
  u += 0x7fffu + ((u >> 16) & 1u);          // round-to-nearest-even
  unsigned short h = (unsigned short)(u >> 16);
  union { unsigned short us; __bf16 b; } cv;
  cv.us = h;
  return cv.b;
}

// ---------------------------------------------------------------------------
// WMMA GEMM: C[M,N] = A[M,K](bf16,row-major) * W(bf16, PRE-TRANSPOSED [N,K])
//            + bias[N], f32 accumulate.
// Block = 128 threads (4 waves), tile 128x64, K-step 32.
// Per wave: 32 rows x 64 cols = 2 A-frags x 4 B-frags = 8 WMMAs per K step.
// Software pipeline: tile kt+1 global->regs issued before the barrier and the
// WMMAs of tile kt; regs->LDS(ping-pong) after compute. One barrier per step.
// ---------------------------------------------------------------------------
__global__ __launch_bounds__(128) void gemm_bf16_wmma(
    const __bf16* __restrict__ A, const __bf16* __restrict__ Wt,
    const float* __restrict__ bias, float* __restrict__ C,
    int M, int K, int N)
{
  __shared__ __align__(16) __bf16 As[2][128][40];  // 128 rows x 32 K (+8 pad)
  __shared__ __align__(16) __bf16 Bs[2][64][40];   // 64 N-rows x 32 K (+8 pad)

  const int tid  = threadIdx.x;
  const int lane = tid & 31;
  const int wave = tid >> 5;
  const int n0 = blockIdx.x * 64;
  const int m0 = blockIdx.y * 128;

  v8f acc[2][4];
#pragma unroll
  for (int mt = 0; mt < 2; ++mt)
#pragma unroll
    for (int t = 0; t < 4; ++t)
#pragma unroll
      for (int i = 0; i < 8; ++i) acc[mt][t][i] = 0.0f;

  const int ar = tid;               // A row 0..127 (64B per thread)
  const int br = tid >> 1;          // B n-row 0..63
  const int bc = (tid & 1) * 16;    // B k-chunk 0 / 16

  const uint4* arow = (const uint4*)(A + (size_t)(m0 + ar) * K);   // [K/8]
  const uint4* brow = (const uint4*)(Wt + (size_t)(n0 + br) * K + bc);

  const int nk = K >> 5;            // K/32 steps; 8 halves per uint4 -> 4/step

  // Prologue: stage tile 0 into LDS buffer 0
  {
    uint4 ta0 = arow[0], ta1 = arow[1], ta2 = arow[2], ta3 = arow[3];
    uint4 tb0 = brow[0], tb1 = brow[1];
    uint4* adst = (uint4*)(&As[0][ar][0]);
    uint4* bdst = (uint4*)(&Bs[0][br][bc]);
    adst[0] = ta0; adst[1] = ta1; adst[2] = ta2; adst[3] = ta3;
    bdst[0] = tb0; bdst[1] = tb1;
  }

  for (int kt = 0; kt < nk; ++kt) {
    const int cur = kt & 1;
    const bool have_next = (kt + 1) < nk;

    // Issue next tile's global loads into named temporaries (distinct VGPRs,
    // latency hides behind this tile's fragment loads + WMMAs).
    uint4 ta0, ta1, ta2, ta3, tb0, tb1;
    if (have_next) {
      const uint4* an = arow + (size_t)(kt + 1) * 4;
      const uint4* bn = brow + (size_t)(kt + 1) * 4;
      ta0 = an[0]; ta1 = an[1]; ta2 = an[2]; ta3 = an[3];
      tb0 = bn[0]; tb1 = bn[1];
    }
    if (kt + 2 < nk) {  // emits global_prefetch_b8 two tiles ahead
      __builtin_prefetch(arow + (size_t)(kt + 2) * 4, 0, 3);
      __builtin_prefetch(brow + (size_t)(kt + 2) * 4, 0, 3);
    }

    __syncthreads();   // LDS buffer `cur` fully written

    // A fragments: 16-bit A 16x32 layout (lanes 0-15: K0-7|16-23; 16-31: +8)
    const int ka = (lane >> 4) << 3;          // 0 or 8
    v16bf afrag[2];
#pragma unroll
    for (int mt = 0; mt < 2; ++mt) {
      const int am = (wave << 5) + (mt << 4) + (lane & 15);
      v8bf a_lo = *(const v8bf*)(&As[cur][am][ka]);
      v8bf a_hi = *(const v8bf*)(&As[cur][am][16 + ka]);
      afrag[mt] = __builtin_shufflevector(a_lo, a_hi,
          0, 1, 2, 3, 4, 5, 6, 7, 8, 9, 10, 11, 12, 13, 14, 15);
    }

#pragma unroll
    for (int t = 0; t < 4; ++t) {
      // B fragment: lanes 0-15 hold K0-7|16-23 of col n; lanes 16-31: +8
      const int bn = (t << 4) + (lane & 15);
      const int kb = (lane >> 4) << 3;        // 0 or 8
      v8bf b_lo = *(const v8bf*)(&Bs[cur][bn][kb]);
      v8bf b_hi = *(const v8bf*)(&Bs[cur][bn][16 + kb]);
      v16bf bfrag = __builtin_shufflevector(b_lo, b_hi,
          0, 1, 2, 3, 4, 5, 6, 7, 8, 9, 10, 11, 12, 13, 14, 15);
      acc[0][t] = __builtin_amdgcn_wmma_f32_16x16x32_bf16(
          false, afrag[0], false, bfrag, (short)0, acc[0][t], false, false);
      acc[1][t] = __builtin_amdgcn_wmma_f32_16x16x32_bf16(
          false, afrag[1], false, bfrag, (short)0, acc[1][t], false, false);
    }

    // Store next tile into the other buffer. Safe without extra barrier:
    // buffer cur^1 was last *read* before the barrier above.
    if (have_next) {
      uint4* adst = (uint4*)(&As[cur ^ 1][ar][0]);
      uint4* bdst = (uint4*)(&Bs[cur ^ 1][br][bc]);
      adst[0] = ta0; adst[1] = ta1; adst[2] = ta2; adst[3] = ta3;
      bdst[0] = tb0; bdst[1] = tb1;
    }
  }

  // C/D layout: VGPR r <-> M=r (lanes 0-15) / M=r+8 (lanes 16-31); N = lane&15
  const int cl = lane & 15;
#pragma unroll
  for (int mt = 0; mt < 2; ++mt) {
    const int rbase = m0 + (wave << 5) + (mt << 4) + ((lane >> 4) << 3);
#pragma unroll
    for (int t = 0; t < 4; ++t) {
      const int col = n0 + (t << 4) + cl;
      const float bv = bias[col];
#pragma unroll
      for (int r = 0; r < 8; ++r)
        C[(size_t)(rbase + r) * N + col] = acc[mt][t][r] + bv;
    }
  }
}

// ---------------------------------------------------------------------------
// Weight conversion: f32 [K,N] row-major -> bf16 TRANSPOSED [N,K].
// LDS-tiled 32x32 transpose so both global read and write are coalesced.
// ---------------------------------------------------------------------------
__global__ __launch_bounds__(256) void cvt_transpose_bf16_kernel(
    const float* __restrict__ src, __bf16* __restrict__ dst, int K, int N)
{
  __shared__ float tile[32][33];
  const int kt = blockIdx.y * 32;
  const int nt = blockIdx.x * 32;
  const int tx = threadIdx.x & 31;
  const int ty = threadIdx.x >> 5;         // 0..7
#pragma unroll
  for (int i = 0; i < 32; i += 8)
    tile[ty + i][tx] = src[(size_t)(kt + ty + i) * N + nt + tx];
  __syncthreads();
#pragma unroll
  for (int i = 0; i < 32; i += 8)
    dst[(size_t)(nt + ty + i) * K + kt + tx] = f2bf(tile[tx][ty + i]);
}

// ---------------------------------------------------------------------------
// out = LayerNorm(a + b) * g + beta ; writes f32 and bf16. One row per block.
// In-place safe for `a` (values cached in registers before write).
// ---------------------------------------------------------------------------
__global__ __launch_bounds__(256) void add_ln_kernel(
    const float* __restrict__ a, const float* __restrict__ b,
    const float* __restrict__ g, const float* __restrict__ beta,
    float* __restrict__ out, __bf16* __restrict__ outbf)
{
  __shared__ float red[256];
  __shared__ float red2[256];
  const int row = blockIdx.x;
  const int t = threadIdx.x;
  float x[3];
  float s = 0.f, s2 = 0.f;
#pragma unroll
  for (int i = 0; i < 3; ++i) {
    const int c = t + i * 256;
    const float v = a[(size_t)row * HID_ + c] + b[(size_t)row * HID_ + c];
    x[i] = v; s += v; s2 += v * v;
  }
  red[t] = s; red2[t] = s2;
  __syncthreads();
  for (int off = 128; off > 0; off >>= 1) {
    if (t < off) { red[t] += red[t + off]; red2[t] += red2[t + off]; }
    __syncthreads();
  }
  const float mean = red[0] * (1.0f / HID_);
  const float var  = red2[0] * (1.0f / HID_) - mean * mean;
  const float rs   = rsqrtf(var + EPS_);
#pragma unroll
  for (int i = 0; i < 3; ++i) {
    const int c = t + i * 256;
    const float y = (x[i] - mean) * rs * g[c] + beta[c];
    out[(size_t)row * HID_ + c] = y;
    outbf[(size_t)row * HID_ + c] = f2bf(y);
  }
}

__global__ void cvt_bf16_kernel(const float* __restrict__ src,
                                __bf16* __restrict__ dst, int n)
{
  const int i = blockIdx.x * blockDim.x + threadIdx.x;
  if (i < n) dst[i] = f2bf(src[i]);
}

__global__ void gelu_bf16_kernel(const float* __restrict__ src,
                                 __bf16* __restrict__ dst, int n)
{
  const int i = blockIdx.x * blockDim.x + threadIdx.x;
  if (i < n) {
    const float v = src[i];
    const float y = 0.5f * v * (1.0f + erff(v * 0.70710678118654752f));
    dst[i] = f2bf(y);
  }
}

// ---------------------------------------------------------------------------
// Local attention: for query s, head h: scores over [G global cols | 513 band]
// with band masked to local_ok & in-range (NEG elsewhere); softmax; PV.
// One (s,h) per 64-thread block.
// ---------------------------------------------------------------------------
__global__ __launch_bounds__(64) void local_attn_kernel(
    const float* __restrict__ Q, const float* __restrict__ K,
    const float* __restrict__ V, const int* __restrict__ amask,
    const int* __restrict__ gmask, float* __restrict__ OUT)
{
  __shared__ float qs[DH_];
  __shared__ float sc[NKEY_];
  __shared__ float red[64];
  __shared__ float bcast;
  const int s = blockIdx.x;
  const int h = blockIdx.y;
  const int t = threadIdx.x;
  const int hoff = h * DH_;
  qs[t] = Q[(size_t)s * HID_ + hoff + t] * 0.125f;   // 1/sqrt(64)
  __syncthreads();

  float lmax = -1.0e30f;
  for (int i = t; i < NKEY_; i += 64) {
    float sv;
    if (i < G_) {                                    // global columns: unmasked
      const float* kp = K + (size_t)i * HID_ + hoff;
      float d = 0.f;
      for (int c = 0; c < DH_; ++c) d += qs[c] * kp[c];
      sv = d;
    } else {
      const int j = s - W1_ + (i - G_);
      const bool ok = (j >= 0) && (j < S_) && (amask[j] > 0) && (gmask[j] == 0);
      if (ok) {
        const float* kp = K + (size_t)j * HID_ + hoff;
        float d = 0.f;
        for (int c = 0; c < DH_; ++c) d += qs[c] * kp[c];
        sv = d;
      } else {
        sv = NEGV_;
      }
    }
    sc[i] = sv;
    lmax = fmaxf(lmax, sv);
  }
  red[t] = lmax;
  __syncthreads();
  if (t == 0) {
    float m = red[0];
    for (int i = 1; i < 64; ++i) m = fmaxf(m, red[i]);
    bcast = m;
  }
  __syncthreads();
  const float smax = bcast;
  float lsum = 0.f;
  for (int i = t; i < NKEY_; i += 64) {
    const float e = expf(sc[i] - smax);
    sc[i] = e;
    lsum += e;
  }
  red[t] = lsum;
  __syncthreads();
  if (t == 0) {
    float ss = 0.f;
    for (int i = 0; i < 64; ++i) ss += red[i];
    bcast = 1.0f / ss;
  }
  __syncthreads();
  const float inv = bcast;

  // dim t: out[d] = sum_i p_i * V[j_i][d]
  float accv = 0.f;
  for (int i = 0; i < NKEY_; ++i) {
    const float p = sc[i];
    const int j = (i < G_) ? i : (s - W1_ + (i - G_));
    if (j >= 0 && j < S_) accv += p * V[(size_t)j * HID_ + hoff + t];
  }
  OUT[(size_t)s * HID_ + hoff + t] = accv * inv;
}

// ---------------------------------------------------------------------------
// Global attention: query row g (<G), head h: full softmax over S keys (keep
// mask), overwrites OUT rows < G. One (g,h) per 256-thread block.
// ---------------------------------------------------------------------------
__global__ __launch_bounds__(256) void global_attn_kernel(
    const float* __restrict__ GQ, const float* __restrict__ GK,
    const float* __restrict__ GV, const int* __restrict__ amask,
    float* __restrict__ OUT)
{
  __shared__ float qs[DH_];
  __shared__ float sc[S_];
  __shared__ float red[256];
  __shared__ float pacc[4][DH_];
  __shared__ float bcast;
  const int g = blockIdx.x;
  const int h = blockIdx.y;
  const int t = threadIdx.x;
  const int hoff = h * DH_;
  if (t < DH_) qs[t] = GQ[(size_t)g * HID_ + hoff + t] * 0.125f;
  __syncthreads();

  float lmax = -1.0e30f;
  for (int i = t; i < S_; i += 256) {
    float sv;
    if (amask[i] > 0) {
      const float* kp = GK + (size_t)i * HID_ + hoff;
      float d = 0.f;
      for (int c = 0; c < DH_; ++c) d += qs[c] * kp[c];
      sv = d;
    } else {
      sv = NEGV_;
    }
    sc[i] = sv;
    lmax = fmaxf(lmax, sv);
  }
  red[t] = lmax;
  __syncthreads();
  if (t == 0) {
    float m = red[0];
    for (int i = 1; i < 256; ++i) m = fmaxf(m, red[i]);
    bcast = m;
  }
  __syncthreads();
  const float smax = bcast;
  float lsum = 0.f;
  for (int i = t; i < S_; i += 256) {
    const float e = expf(sc[i] - smax);
    sc[i] = e;
    lsum += e;
  }
  red[t] = lsum;
  __syncthreads();
  if (t == 0) {
    float ss = 0.f;
    for (int i = 0; i < 256; ++i) ss += red[i];
    bcast = 1.0f / ss;
  }
  __syncthreads();
  const float inv = bcast;

  const int part = t >> 6;
  const int d = t & 63;
  float accv = 0.f;
  for (int i = part * 512; i < part * 512 + 512; ++i)
    accv += sc[i] * GV[(size_t)i * HID_ + hoff + d];
  pacc[part][d] = accv;
  __syncthreads();
  if (t < DH_) {
    const float tot = (pacc[0][t] + pacc[1][t] + pacc[2][t] + pacc[3][t]) * inv;
    OUT[(size_t)g * HID_ + hoff + t] = tot;
  }
}

// logits[s,c] = H[s,:] . cls_w[:,c] + cls_b[c]
__global__ void classifier_kernel(const float* __restrict__ H,
                                  const float* __restrict__ w,
                                  const float* __restrict__ b,
                                  float* __restrict__ logits)
{
  const int idx = blockIdx.x * blockDim.x + threadIdx.x;
  if (idx >= S_ * 2) return;
  const int s = idx >> 1, c = idx & 1;
  float d = b[c];
  const float* hp = H + (size_t)s * HID_;
  for (int k = 0; k < HID_; ++k) d += hp[k] * w[k * 2 + c];
  logits[idx] = d;
}

// p = softmax(logits)[:,1] -> reshape(32,64)[1:, :-1] -> prod/sum/start/end +
// stable descending argsort of sums.
__global__ __launch_bounds__(32) void postprocess_kernel(
    const float* __restrict__ logits, float* __restrict__ outp)
{
  __shared__ float sums[NR_];
  const int t = threadIdx.x;
  if (t < NR_) {
    const int r = t + 1;
    float prod = 1.f, sum = 0.f, pstart = 0.f, pend = 0.f;
    for (int c = 0; c < 63; ++c) {
      const int s = r * 64 + c;
      const float p = 1.0f / (1.0f + expf(logits[s * 2] - logits[s * 2 + 1]));
      prod *= p;
      sum += p;
      if (c == 0)  pstart = p;
      if (c == 62) pend = p;
    }
    outp[0 * NR_ + t] = prod;
    outp[1 * NR_ + t] = sum;
    outp[2 * NR_ + t] = pstart;
    outp[3 * NR_ + t] = pend;
    sums[t] = sum;
  }
  __syncthreads();
  if (t == 0) {
    unsigned used = 0;
    for (int pos = 0; pos < NR_; ++pos) {
      int best = -1;
      for (int i = 0; i < NR_; ++i) {
        if (used & (1u << i)) continue;
        if (best < 0 || sums[i] > sums[best]) best = i;   // stable on ties
      }
      used |= (1u << best);
      outp[4 * NR_ + pos] = (float)best;
    }
  }
}

// ---------------------------------------------------------------------------
extern "C" void kernel_launch(void* const* d_in, const int* in_sizes, int n_in,
                              void* d_out, int out_size, void* d_ws, size_t ws_size,
                              hipStream_t stream)
{
  (void)in_sizes; (void)n_in; (void)out_size; (void)ws_size;

  const float* embeds = (const float*)d_in[0];
  const int*   amask  = (const int*)d_in[1];
  const int*   gmask  = (const int*)d_in[2];
  const float* pos    = (const float*)d_in[3];
  const float* eg     = (const float*)d_in[4];
  const float* eb     = (const float*)d_in[5];
  const float* q_w  = (const float*)d_in[6];  const float* q_b  = (const float*)d_in[7];
  const float* k_w  = (const float*)d_in[8];  const float* k_b  = (const float*)d_in[9];
  const float* v_w  = (const float*)d_in[10]; const float* v_b  = (const float*)d_in[11];
  const float* gq_w = (const float*)d_in[12]; const float* gq_b = (const float*)d_in[13];
  const float* gk_w = (const float*)d_in[14]; const float* gk_b = (const float*)d_in[15];
  const float* gv_w = (const float*)d_in[16]; const float* gv_b = (const float*)d_in[17];
  const float* o_w  = (const float*)d_in[18]; const float* o_b  = (const float*)d_in[19];
  const float* ln1g = (const float*)d_in[20]; const float* ln1b = (const float*)d_in[21];
  const float* f1_w = (const float*)d_in[22]; const float* f1_b = (const float*)d_in[23];
  const float* f2_w = (const float*)d_in[24]; const float* f2_b = (const float*)d_in[25];
  const float* ln2g = (const float*)d_in[26]; const float* ln2b = (const float*)d_in[27];
  const float* cls_w = (const float*)d_in[28]; const float* cls_b = (const float*)d_in[29];

  // ---- workspace carve (~87 MB) ----
  char* p = (char*)d_ws;
  auto carve = [&](size_t bytes) -> char* {
    char* r = p;
    p += (bytes + 255) & ~(size_t)255;
    return r;
  };
  float*  H     = (float*)carve((size_t)S_ * HID_ * 4);
  __bf16* Hbf   = (__bf16*)carve((size_t)S_ * HID_ * 2);
  __bf16* Wbf   = (__bf16*)carve((size_t)HID_ * FFN_ * 2);   // largest weight
  float*  Qb    = (float*)carve((size_t)S_ * HID_ * 4);
  float*  Kb    = (float*)carve((size_t)S_ * HID_ * 4);
  float*  Vb    = (float*)carve((size_t)S_ * HID_ * 4);
  float*  ATT   = (float*)carve((size_t)S_ * HID_ * 4);
  __bf16* ATTbf = (__bf16*)carve((size_t)S_ * HID_ * 2);
  float*  F1    = (float*)carve((size_t)S_ * FFN_ * 4);
  __bf16* F1bf  = (__bf16*)carve((size_t)S_ * FFN_ * 2);
  float*  TMP   = (float*)carve((size_t)S_ * HID_ * 4);

  const int WHH = HID_ * HID_;
  const int WHF = HID_ * FFN_;
  const dim3 gemmBlock(128);
  const dim3 gridH(HID_ / 64, S_ / 128);      // (12, 16)
  const dim3 gridF(FFN_ / 64, S_ / 128);      // (48, 16)
  const dim3 trHH(HID_ / 32, HID_ / 32);      // transpose grids (N/32, K/32)
  const dim3 trHF(FFN_ / 32, HID_ / 32);      // f1: K=HID, N=FFN
  const dim3 trFH(HID_ / 32, FFN_ / 32);      // f2: K=FFN, N=HID
  auto nb = [](int n) { return (n + 255) / 256; };

  // Embedding + LayerNorm
  add_ln_kernel<<<S_, 256, 0, stream>>>(embeds, pos, eg, eb, H, Hbf);

  for (int l = 0; l < NL_; ++l) {
    // Local q/k/v projections (WMMA; weights converted+transposed to [N,K])
    cvt_transpose_bf16_kernel<<<trHH, 256, 0, stream>>>(q_w + (size_t)l * WHH, Wbf, HID_, HID_);
    gemm_bf16_wmma<<<gridH, gemmBlock, 0, stream>>>(Hbf, Wbf, q_b + l * HID_, Qb, S_, HID_, HID_);
    cvt_transpose_bf16_kernel<<<trHH, 256, 0, stream>>>(k_w + (size_t)l * WHH, Wbf, HID_, HID_);
    gemm_bf16_wmma<<<gridH, gemmBlock, 0, stream>>>(Hbf, Wbf, k_b + l * HID_, Kb, S_, HID_, HID_);
    cvt_transpose_bf16_kernel<<<trHH, 256, 0, stream>>>(v_w + (size_t)l * WHH, Wbf, HID_, HID_);
    gemm_bf16_wmma<<<gridH, gemmBlock, 0, stream>>>(Hbf, Wbf, v_b + l * HID_, Vb, S_, HID_, HID_);

    // Banded local attention (+ global columns)
    local_attn_kernel<<<dim3(S_, NH_), 64, 0, stream>>>(Qb, Kb, Vb, amask, gmask, ATT);

    // Global q/k/v projections (reuse Qb/Kb/Vb)
    cvt_transpose_bf16_kernel<<<trHH, 256, 0, stream>>>(gq_w + (size_t)l * WHH, Wbf, HID_, HID_);
    gemm_bf16_wmma<<<gridH, gemmBlock, 0, stream>>>(Hbf, Wbf, gq_b + l * HID_, Qb, S_, HID_, HID_);
    cvt_transpose_bf16_kernel<<<trHH, 256, 0, stream>>>(gk_w + (size_t)l * WHH, Wbf, HID_, HID_);
    gemm_bf16_wmma<<<gridH, gemmBlock, 0, stream>>>(Hbf, Wbf, gk_b + l * HID_, Kb, S_, HID_, HID_);
    cvt_transpose_bf16_kernel<<<trHH, 256, 0, stream>>>(gv_w + (size_t)l * WHH, Wbf, HID_, HID_);
    gemm_bf16_wmma<<<gridH, gemmBlock, 0, stream>>>(Hbf, Wbf, gv_b + l * HID_, Vb, S_, HID_, HID_);

    // Global attention overwrites rows < G
    global_attn_kernel<<<dim3(G_, NH_), 256, 0, stream>>>(Qb, Kb, Vb, amask, ATT);

    // Output projection + residual LN1
    cvt_bf16_kernel<<<nb(S_ * HID_), 256, 0, stream>>>(ATT, ATTbf, S_ * HID_);
    cvt_transpose_bf16_kernel<<<trHH, 256, 0, stream>>>(o_w + (size_t)l * WHH, Wbf, HID_, HID_);
    gemm_bf16_wmma<<<gridH, gemmBlock, 0, stream>>>(ATTbf, Wbf, o_b + l * HID_, TMP, S_, HID_, HID_);
    add_ln_kernel<<<S_, 256, 0, stream>>>(H, TMP, ln1g + l * HID_, ln1b + l * HID_, H, Hbf);

    // FFN
    cvt_transpose_bf16_kernel<<<trHF, 256, 0, stream>>>(f1_w + (size_t)l * WHF, Wbf, HID_, FFN_);
    gemm_bf16_wmma<<<gridF, gemmBlock, 0, stream>>>(Hbf, Wbf, f1_b + l * FFN_, F1, S_, HID_, FFN_);
    gelu_bf16_kernel<<<nb(S_ * FFN_), 256, 0, stream>>>(F1, F1bf, S_ * FFN_);
    cvt_transpose_bf16_kernel<<<trFH, 256, 0, stream>>>(f2_w + (size_t)l * WHF, Wbf, FFN_, HID_);
    gemm_bf16_wmma<<<gridH, gemmBlock, 0, stream>>>(F1bf, Wbf, f2_b + l * HID_, TMP, S_, FFN_, HID_);
    add_ln_kernel<<<S_, 256, 0, stream>>>(H, TMP, ln2g + l * HID_, ln2b + l * HID_, H, Hbf);
  }

  // Classifier + post-processing (outputs concatenated flat into d_out)
  float* outF = (float*)d_out;
  classifier_kernel<<<nb(S_ * 2), 256, 0, stream>>>(H, cls_w, cls_b, outF);
  postprocess_kernel<<<1, 32, 0, stream>>>(outF, outF + S_ * 2);
}
